// EGNN_Sparse_Network_32057635897677
// MI455X (gfx1250) — matrix operand order repaired and checked by reading, not compiled, not run on hardware
//
#include <hip/hip_runtime.h>
#include <hip/hip_bf16.h>

typedef __attribute__((ext_vector_type(16))) _Float16 v16h;
typedef __attribute__((ext_vector_type(8)))  _Float16 v8h;
typedef __attribute__((ext_vector_type(8)))  float    v8f;

// ---- network shape constants (fixed by the reference) ----
constexpr int XDIM = 67;           // 64 feats + 3 coors
constexpr int K1   = 160;          // edge_in 133 padded to 5*32
constexpr int KT1  = 5;
constexpr int N1T  = 18;           // 266 padded to 288 = 18 tiles of 16
constexpr int K2   = 288;          // GEMM2 K = 266 padded (== N1T*16)
constexpr int KT2  = 9;
constexpr int K3   = 96;           // node_in 80 padded to 3*32
constexpr int KT3  = 3;
constexpr int N3T  = 8;            // 128 hidden
constexpr int K4   = 128;
constexpr int KT4  = 4;
constexpr int N4T  = 4;            // 64 out

// f16 col-major weight tile sizes (per layer, in halves)
constexpr int S1 = 288 * K1;       // edge_w1^T  [288][160]
constexpr int S2 = 16  * K2;       // edge_w2^T  [16][288]
constexpr int S3 = 128 * K3;       // node_w1^T  [128][96]
constexpr int S4 = 64  * K4;       // node_w2^T  [64][128]
constexpr int PER = S1 + S2 + S3 + S4;

// SiLU with fast hardware reciprocal (v_rcp_f32) instead of IEEE divide:
// the full-precision divide lowered to a ~10-op div_scale/fma chain per
// element, on the critical path between WMMA tiles.  exp already lowers to
// a single v_exp_f32.
__device__ __forceinline__ float silu(float v) {
    return v * __builtin_amdgcn_rcpf(1.0f + __expf(-v));
}

__device__ __forceinline__ v8f wmma_f16(v16h a, v16h b, v8f c) {
    return __builtin_amdgcn_wmma_f32_16x16x32_f16(false, a, false, b, (short)0, c, false, false);
}

// Load a WMMA A-fragment (16x32 f16) from an LDS row-major matrix with row
// stride `stride` halves.  A layout: lanes 0-15 hold row (lane) with
// K {0..15}; lanes 16-31 hold row (lane-16) with K {16..31}.
__device__ __forceinline__ v16h load_a_frag(const _Float16* base, int row, int stride,
                                            int kt, int hi) {
    union { v16h v; v8h h[2]; } f;
    const _Float16* p = base + row * stride + kt * 32 + hi * 16;
    f.h[0] = *reinterpret_cast<const v8h*>(p);
    f.h[1] = *reinterpret_cast<const v8h*>(p + 8);
    return f.v;
}

// =====================  weight preparation  =====================
// f32 row-major [K][N]  ->  f16 col-major padded [NPAD][KPAD], zeros in pad.
__global__ void prep_weights_kernel(const float* __restrict__ ew1, const float* __restrict__ ew2,
                                    const float* __restrict__ nw1, const float* __restrict__ nw2,
                                    _Float16* __restrict__ w1t, _Float16* __restrict__ w2t,
                                    _Float16* __restrict__ nw1t, _Float16* __restrict__ nw2t) {
    int id = blockIdx.x * blockDim.x + threadIdx.x;
    if (id >= 3 * PER) return;
    int l = id / PER;
    int r = id - l * PER;
    if (r < S1) {
        int n = r / K1, k = r - n * K1;
        float v = (n < 266 && k < 133) ? ew1[(l * 133 + k) * 266 + n] : 0.0f;
        w1t[(size_t)l * S1 + r] = (_Float16)v;
    } else if ((r -= S1) < S2) {
        int n = r / K2, k = r - n * K2;
        float v = (k < 266) ? ew2[(l * 266 + k) * 16 + n] : 0.0f;
        w2t[(size_t)l * S2 + r] = (_Float16)v;
    } else if ((r -= S2) < S3) {
        int n = r / K3, k = r - n * K3;
        float v = (k < 80) ? nw1[(l * 80 + k) * 128 + n] : 0.0f;
        nw1t[(size_t)l * S3 + r] = (_Float16)v;
    } else {
        r -= S3;
        int n = r / K4, k = r - n * K4;
        nw2t[(size_t)l * S4 + r] = (_Float16)nw2[(l * 128 + k) * 64 + n];
    }
}

__global__ void zero_accum_kernel(float* __restrict__ m_i, float* __restrict__ coor_wi,
                                  float* __restrict__ coor_ri, int N) {
    int id = blockIdx.x * blockDim.x + threadIdx.x;
    if (id < N * 16) m_i[id] = 0.0f;
    if (id < N)      coor_wi[id] = 0.0f;
    if (id < N * 3)  coor_ri[id] = 0.0f;
}

// =====================  edge kernel  =====================
// 1 wave = tile of 16 edges.  Fused: gather -> GEMM1(133->266) -> SiLU ->
// GEMM2(266->16) -> SiLU -> coor MLP + atomic scatters.
__global__ __launch_bounds__(128) void egnn_edge_kernel(
    const float* __restrict__ x, const int* __restrict__ ei, const float* __restrict__ eattr,
    const _Float16* __restrict__ w1t, const float* __restrict__ b1,
    const _Float16* __restrict__ w2t, const float* __restrict__ b2,
    const float* __restrict__ cw1, const float* __restrict__ cb1,
    const float* __restrict__ cw2, const float* __restrict__ cb2,
    float* __restrict__ m_i, float* __restrict__ coor_wi, float* __restrict__ coor_ri,
    int E) {
    __shared__ _Float16 sA1[4][16 * K1];   // 20480 B  staged edge_in (f16)
    __shared__ _Float16 sA2[4][16 * K2];   // 36864 B  staged hidden (f16)
    __shared__ float    sM [4][16 * 16];   //  4096 B  m_ij (f32) for coor MLP
    __shared__ float    sRel[4][16 * 4];   //  1024 B  rel_coors + dist
    __shared__ int      sDst[4][16];       //   256 B

    const int wave = threadIdx.x >> 5;
    const int lane = threadIdx.x & 31;
    const int ln   = lane & 15;
    const int hi   = lane >> 4;
    const int tile = blockIdx.x * 4 + wave;
    const bool valid = tile < (E >> 4);

    _Float16* A1 = sA1[wave];
    _Float16* A2 = sA2[wave];
    float mv[8];

    if (valid) {
        const int e  = tile * 16 + ln;
        const int sI = ei[e];
        const int dI = ei[E + e];
        float rx = x[(size_t)sI * XDIM + 64] - x[(size_t)dI * XDIM + 64];
        float ry = x[(size_t)sI * XDIM + 65] - x[(size_t)dI * XDIM + 65];
        float rz = x[(size_t)sI * XDIM + 66] - x[(size_t)dI * XDIM + 66];
        float dist = rx * rx + ry * ry + rz * rz;
        if (hi == 0) {
            sRel[wave][ln * 4 + 0] = rx; sRel[wave][ln * 4 + 1] = ry;
            sRel[wave][ln * 4 + 2] = rz; sRel[wave][ln * 4 + 3] = dist;
            sDst[wave][ln] = dI;
        }
        // gather: cols 0-63 = feats[dst] (hi=0), 64-127 = feats[src] (hi=1)
        const float* frow = x + (size_t)(hi ? sI : dI) * XDIM;
        _Float16* arow = A1 + ln * K1 + hi * 64;
        #pragma unroll
        for (int c = 0; c < 64; c += 8) {
            v8h v;
            #pragma unroll
            for (int j = 0; j < 8; ++j) v[j] = (_Float16)frow[c + j];
            *reinterpret_cast<v8h*>(arow + c) = v;
        }
        if (hi) {   // cols 128-131 attr, 132 dist, 133-159 zero pad
            _Float16* ex = A1 + ln * K1 + 128;
            ex[0] = (_Float16)eattr[(size_t)e * 4 + 0];
            ex[1] = (_Float16)eattr[(size_t)e * 4 + 1];
            ex[2] = (_Float16)eattr[(size_t)e * 4 + 2];
            ex[3] = (_Float16)eattr[(size_t)e * 4 + 3];
            ex[4] = (_Float16)dist;
            #pragma unroll
            for (int c = 5; c < 32; ++c) ex[c] = (_Float16)0.0f;
        }
    }
    __syncthreads();

    if (valid) {  // ---- GEMM1: (16x160) x (160x288) ----
        v16h a1[KT1];
        #pragma unroll
        for (int kt = 0; kt < KT1; ++kt) a1[kt] = load_a_frag(A1, ln, K1, kt, hi);
        for (int nt = 0; nt < N1T; ++nt) {
            const int ncol = nt * 16 + ln;
            float bias = (ncol < 266) ? b1[ncol] : 0.0f;
            v8f acc = {bias, bias, bias, bias, bias, bias, bias, bias};
            #pragma unroll
            for (int kt = 0; kt < KT1; ++kt) {
                v16h b = *reinterpret_cast<const v16h*>(w1t + (size_t)ncol * K1 + kt * 32 + hi * 16);
                acc = wmma_f16(a1[kt], b, acc);
            }
            #pragma unroll
            for (int r = 0; r < 8; ++r)   // D elem (row=r+8*hi, col=ncol)
                A2[(r + hi * 8) * K2 + ncol] = (_Float16)silu(acc[r]);
        }
    }
    __syncthreads();

    if (valid) {  // ---- GEMM2: (16x288) x (288x16) ----
        v16h a2[KT2];
        #pragma unroll
        for (int kt = 0; kt < KT2; ++kt) a2[kt] = load_a_frag(A2, ln, K2, kt, hi);
        float bias = b2[ln];
        v8f acc = {bias, bias, bias, bias, bias, bias, bias, bias};
        #pragma unroll
        for (int kt = 0; kt < KT2; ++kt) {
            v16h b = *reinterpret_cast<const v16h*>(w2t + (size_t)ln * K2 + kt * 32 + hi * 16);
            acc = wmma_f16(a2[kt], b, acc);
        }
        #pragma unroll
        for (int r = 0; r < 8; ++r) {
            mv[r] = silu(acc[r]);
            sM[wave][(r + hi * 8) * 16 + ln] = mv[r];
        }
    }
    __syncthreads();

    if (valid) {
        // scatter m_i (segment sum over dst)
        #pragma unroll
        for (int r = 0; r < 8; ++r) {
            int d = sDst[wave][r + hi * 8];
            __hip_atomic_fetch_add(&m_i[(size_t)d * 16 + ln], mv[r],
                                   __ATOMIC_RELAXED, __HIP_MEMORY_SCOPE_AGENT);
        }
        // coor MLP (16->64->1) on VALU, 1 lane per edge
        if (hi == 0) {
            const float* mrow = &sM[wave][ln * 16];
            float cw = cb2[0];
            for (int j = 0; j < 64; ++j) {
                float a = cb1[j];
                #pragma unroll
                for (int k = 0; k < 16; ++k) a += mrow[k] * cw1[k * 64 + j];
                cw += silu(a) * cw2[j];
            }
            int d = sDst[wave][ln];
            __hip_atomic_fetch_add(&coor_wi[d], cw, __ATOMIC_RELAXED, __HIP_MEMORY_SCOPE_AGENT);
            __hip_atomic_fetch_add(&coor_ri[(size_t)d * 3 + 0], sRel[wave][ln * 4 + 0],
                                   __ATOMIC_RELAXED, __HIP_MEMORY_SCOPE_AGENT);
            __hip_atomic_fetch_add(&coor_ri[(size_t)d * 3 + 1], sRel[wave][ln * 4 + 1],
                                   __ATOMIC_RELAXED, __HIP_MEMORY_SCOPE_AGENT);
            __hip_atomic_fetch_add(&coor_ri[(size_t)d * 3 + 2], sRel[wave][ln * 4 + 2],
                                   __ATOMIC_RELAXED, __HIP_MEMORY_SCOPE_AGENT);
        }
    }
}

// =====================  node kernel  =====================
// 1 wave = 16 nodes: node_in(80) -> 128 -> 64 via WMMA + residual; coors out.
__global__ __launch_bounds__(128) void egnn_node_kernel(
    const float* __restrict__ x, const float* __restrict__ m_i,
    const float* __restrict__ coor_wi, const float* __restrict__ coor_ri,
    const _Float16* __restrict__ nw1t, const float* __restrict__ nb1,
    const _Float16* __restrict__ nw2t, const float* __restrict__ nb2,
    float* __restrict__ xout, int N) {
    __shared__ _Float16 sA[4][16 * K3];   // 12288 B
    __shared__ _Float16 sH[4][16 * K4];   // 16384 B

    const int wave = threadIdx.x >> 5;
    const int lane = threadIdx.x & 31;
    const int ln   = lane & 15;
    const int hi   = lane >> 4;
    const int tile = blockIdx.x * 4 + wave;
    const bool valid = tile < (N >> 4);
    const int nbase = tile * 16;

    _Float16* A = sA[wave];
    _Float16* H = sH[wave];

    if (valid) {
        const int node = nbase + ln;
        const float* xr = x + (size_t)node * XDIM;
        _Float16* arow = A + ln * K3;
        if (hi == 0) {           // cols 0..47: feats
            #pragma unroll
            for (int c = 0; c < 48; c += 8) {
                v8h v;
                #pragma unroll
                for (int j = 0; j < 8; ++j) v[j] = (_Float16)xr[c + j];
                *reinterpret_cast<v8h*>(arow + c) = v;
            }
        } else {                 // cols 48..63 feats, 64..79 m_i, 80..95 zero
            #pragma unroll
            for (int c = 48; c < 64; c += 8) {
                v8h v;
                #pragma unroll
                for (int j = 0; j < 8; ++j) v[j] = (_Float16)xr[c + j];
                *reinterpret_cast<v8h*>(arow + c) = v;
            }
            const float* mr = m_i + (size_t)node * 16;
            #pragma unroll
            for (int c = 0; c < 16; c += 8) {
                v8h v;
                #pragma unroll
                for (int j = 0; j < 8; ++j) v[j] = (_Float16)mr[c + j];
                *reinterpret_cast<v8h*>(arow + 64 + c) = v;
            }
            v8h z;
            #pragma unroll
            for (int j = 0; j < 8; ++j) z[j] = (_Float16)0.0f;
            *reinterpret_cast<v8h*>(arow + 80) = z;
            *reinterpret_cast<v8h*>(arow + 88) = z;
        }
    }
    __syncthreads();

    if (valid) {  // ---- (16x96) x (96x128) ----
        v16h a[KT3];
        #pragma unroll
        for (int kt = 0; kt < KT3; ++kt) a[kt] = load_a_frag(A, ln, K3, kt, hi);
        for (int nt = 0; nt < N3T; ++nt) {
            const int ncol = nt * 16 + ln;
            float bias = nb1[ncol];
            v8f acc = {bias, bias, bias, bias, bias, bias, bias, bias};
            #pragma unroll
            for (int kt = 0; kt < KT3; ++kt) {
                v16h b = *reinterpret_cast<const v16h*>(nw1t + (size_t)ncol * K3 + kt * 32 + hi * 16);
                acc = wmma_f16(a[kt], b, acc);
            }
            #pragma unroll
            for (int r = 0; r < 8; ++r)
                H[(r + hi * 8) * K4 + ncol] = (_Float16)silu(acc[r]);
        }
    }
    __syncthreads();

    if (valid) {  // ---- (16x128) x (128x64) + residual ----
        v16h a2[KT4];
        #pragma unroll
        for (int kt = 0; kt < KT4; ++kt) a2[kt] = load_a_frag(H, ln, K4, kt, hi);
        for (int nt = 0; nt < N4T; ++nt) {
            const int ncol = nt * 16 + ln;
            float bias = nb2[ncol];
            v8f acc = {bias, bias, bias, bias, bias, bias, bias, bias};
            #pragma unroll
            for (int kt = 0; kt < KT4; ++kt) {
                v16h b = *reinterpret_cast<const v16h*>(nw2t + (size_t)ncol * K4 + kt * 32 + hi * 16);
                acc = wmma_f16(a2[kt], b, acc);
            }
            #pragma unroll
            for (int r = 0; r < 8; ++r) {
                size_t o = (size_t)(nbase + r + hi * 8) * XDIM + ncol;
                xout[o] = acc[r] + x[o];   // residual
            }
        }
        if (hi == 0) {   // coors_out = coors + coor_wi * coor_ri
            const int node = nbase + ln;
            float w = coor_wi[node];
            #pragma unroll
            for (int i = 0; i < 3; ++i)
                xout[(size_t)node * XDIM + 64 + i] =
                    x[(size_t)node * XDIM + 64 + i] + w * coor_ri[(size_t)node * 3 + i];
        }
    }
}

// =====================  host-side launch  =====================
extern "C" void kernel_launch(void* const* d_in, const int* in_sizes, int n_in,
                              void* d_out, int out_size, void* d_ws, size_t ws_size,
                              hipStream_t stream) {
    (void)n_in; (void)out_size; (void)ws_size;
    const float* x     = (const float*)d_in[0];
    const int*   ei    = (const int*)d_in[1];    // JAX default x64-off: int32
    const float* eattr = (const float*)d_in[2];
    const float* ew1   = (const float*)d_in[3];
    const float* eb1   = (const float*)d_in[4];
    const float* ew2   = (const float*)d_in[5];
    const float* eb2   = (const float*)d_in[6];
    const float* cw1   = (const float*)d_in[7];
    const float* cb1   = (const float*)d_in[8];
    const float* cw2   = (const float*)d_in[9];
    const float* cb2   = (const float*)d_in[10];
    const float* nw1   = (const float*)d_in[11];
    const float* nb1   = (const float*)d_in[12];
    const float* nw2   = (const float*)d_in[13];
    const float* nb2   = (const float*)d_in[14];

    const int N = in_sizes[0] / XDIM;
    const int E = in_sizes[1] / 2;

    // workspace layout
    _Float16* w1t  = (_Float16*)d_ws;
    _Float16* w2t  = w1t  + (size_t)3 * S1;
    _Float16* nw1t = w2t  + (size_t)3 * S2;
    _Float16* nw2t = nw1t + (size_t)3 * S3;
    float* m_i     = (float*)(nw2t + (size_t)3 * S4);
    float* coor_wi = m_i     + (size_t)N * 16;
    float* coor_ri = coor_wi + (size_t)N;
    float* xb0     = coor_ri + (size_t)N * 3;
    float* xb1     = xb0     + (size_t)N * XDIM;

    prep_weights_kernel<<<(3 * PER + 255) / 256, 256, 0, stream>>>(
        ew1, ew2, nw1, nw2, w1t, w2t, nw1t, nw2t);

    for (int l = 0; l < 3; ++l) {
        zero_accum_kernel<<<(N * 16 + 255) / 256, 256, 0, stream>>>(m_i, coor_wi, coor_ri, N);

        const float* xin = (l == 0) ? x : ((l == 1) ? xb0 : xb1);
        float* xo        = (l == 0) ? xb0 : ((l == 1) ? xb1 : (float*)d_out);

        const int etiles = E / 16;
        egnn_edge_kernel<<<(etiles + 3) / 4, 128, 0, stream>>>(
            xin, ei, eattr,
            w1t + (size_t)l * S1, eb1 + (size_t)l * 266,
            w2t + (size_t)l * S2, eb2 + (size_t)l * 16,
            cw1 + (size_t)l * 16 * 64, cb1 + (size_t)l * 64,
            cw2 + (size_t)l * 64, cb2 + l,
            m_i, coor_wi, coor_ri, E);

        const int ntiles = N / 16;
        egnn_node_kernel<<<(ntiles + 3) / 4, 128, 0, stream>>>(
            xin, m_i, coor_wi, coor_ri,
            nw1t + (size_t)l * S3, nb1 + (size_t)l * 128,
            nw2t + (size_t)l * S4, nb2 + (size_t)l * 64,
            xo, N);
    }
}
